// MultiHeadAttention_25031069401450
// MI455X (gfx1250) — matrix operand order repaired and checked by reading, not compiled, not run on hardware
//
#include <hip/hip_runtime.h>
#include <hip/hip_bf16.h>

// ---------------- CDNA5 types ----------------
typedef __attribute__((ext_vector_type(16))) __bf16       v16bf;
typedef __attribute__((ext_vector_type(8)))  float        v8f;
typedef __attribute__((ext_vector_type(4))) unsigned int  u32x4;
typedef __attribute__((ext_vector_type(2))) unsigned int  u32x2;
typedef __attribute__((ext_vector_type(8)))  int          i32x8;
typedef __attribute__((ext_vector_type(4)))  int          i32x4;

union FragAB { v16bf v; u32x4 q[2]; };

#define BDIM 2
#define SEQ  1024
#define DM   1024
#define NH   16
#define DH   64
#define RELN 320   // 2*128+1 = 257 padded to 320 (multiple of 64)

__device__ __forceinline__ unsigned short f2bf(float f) {
  unsigned int u = __float_as_uint(f);
  u += 0x7FFFu + ((u >> 16) & 1u);           // round-to-nearest-even
  return (unsigned short)(u >> 16);
}
__device__ __forceinline__ float bf2f(unsigned short h) {
  return __uint_as_float(((unsigned int)h) << 16);
}

// ---------------- generic fp32 -> bf16 convert ----------------
__global__ void cvt_bf16_kernel(const float* __restrict__ in,
                                unsigned short* __restrict__ out, int n) {
  int i = blockIdx.x * blockDim.x + threadIdx.x;
  if (i < n) out[i] = f2bf(in[i]);
}

// ---------------- rel_k table -> padded bf16 [RELN][64] ----------------
__global__ void pack_relk_kernel(const float* __restrict__ relk,
                                 unsigned short* __restrict__ out) {
  int i = blockIdx.x * blockDim.x + threadIdx.x;
  if (i >= RELN * DH) return;
  int j = i >> 6;
  int d = i & 63;
  out[i] = f2bf(j < 257 ? relk[j * DH + d] : 0.0f);
}

// ---------------- Tensor Data Mover: 2D bf16 tile -> LDS with row padding -------
// Loads tile_rows x 32 bf16 elements (64B rows) from row stride lda_elems into
// LDS at lds_addr with +16B padding after every 64B row (LDS stride = 40 halfs).
// D# per CDNA5 ISA:
//   group0: [1:0]count=1 | [63:32]lds_addr | [120:64]global_addr | [127:126]type=2
//   group1: [17:16]data_size=1(2B) | [20]pad_enable | [24:22]pad_interval=3(16 DW)
//           | [31:25]pad_amount=3(4 DW) | tensor_dim0/1 | tile_dim0=32,
//           tile_dim1=rows | tensor_dim0_stride=lda
// Toolchain here is clang-23 / therock-10.0: 6-arg builtin
//   (u32x4 g0, i32x8 g1, i32x4 g2, i32x4 g3, i32x8 g4, i32 cpol)
#if __has_builtin(__builtin_amdgcn_tensor_load_to_lds)
#define HAVE_TDM 1
__device__ __forceinline__ void tdm_load_tile(unsigned lds_addr, const void* gaddr,
                                              int lda_elems, int tile_rows) {
  unsigned long long ga = (unsigned long long)(uintptr_t)gaddr;
  u32x4 g0;
  g0[0] = 1u;                                             // count=1, user mode
  g0[1] = lds_addr;                                       // LDS byte address
  g0[2] = (unsigned)(ga & 0xFFFFFFFFu);                   // global addr lo
  g0[3] = (unsigned)((ga >> 32) & 0x01FFFFFFu) | (2u << 30);  // addr hi | type=2
  unsigned td0 = (unsigned)lda_elems;                     // tensor dim0 (elements)
  unsigned td1 = 1u << 20;                                // generous dim1 bound
  i32x8 g1;
  g1[0] = (int)((1u << 16) | (1u << 20) | (3u << 22) | (3u << 25));
  g1[1] = (int)((td0 & 0xFFFFu) << 16);                   // tensor_dim0 lo16
  g1[2] = (int)(((td0 >> 16) & 0xFFFFu) | ((td1 & 0xFFFFu) << 16));
  g1[3] = (int)(((td1 >> 16) & 0xFFFFu) | (32u << 16));   // tile_dim0 = 32
  g1[4] = (int)(unsigned)tile_rows;                       // tile_dim1, tile_dim2=0
  g1[5] = (int)td0;                                       // dim0 stride lo32
  g1[6] = 0;                                              // stride hi, dim1 stride
  g1[7] = 0;
  i32x4 z4 = {0, 0, 0, 0};
  i32x8 z8 = {0, 0, 0, 0, 0, 0, 0, 0};
  __builtin_amdgcn_tensor_load_to_lds(g0, g1, z4, z4, z8, 0);
}
#else
#define HAVE_TDM 0
#endif

// ---------------- batched bf16 NT GEMM with WMMA + TDM double buffering --------
// C(MxN) = A(MxK,row-major,lda) * B(NxK,row-major,ldb)^T  (+bias[n])
// grid: (N/64, M/128, nbatch). 256 threads = 8 waves; each wave a 32x32 block
// (2x2 v_wmma_f32_16x16x32_bf16 tiles), K step 32. Tiles staged in LDS by the
// Tensor Data Mover (wave 0 issues DMA; TENSORcnt + barrier synchronize), with
// the next K-tile's DMA overlapping the current tile's WMMA work.
// C batch offset = (z % cMod)*cs1 + (z / cMod)*cs2  (handles head-interleaved ctx).
__global__ void __launch_bounds__(256)
gemm_bf16_nt(const unsigned short* __restrict__ A, long sA, int lda,
             const unsigned short* __restrict__ B, long sB, int ldb,
             void* __restrict__ C, int ldc, long cMod, long cs1, long cs2,
             const float* __restrict__ bias, int K, int out_bf16) {
  __shared__ __align__(16) unsigned short As[2][128 * 40];  // 128x32, stride 40
  __shared__ __align__(16) unsigned short Bs[2][64 * 40];   //  64x32, stride 40

  const int tid  = threadIdx.x;
  const int wave = tid >> 5;
  const int lane = tid & 31;
  const int m0   = (wave >> 1) * 32;
  const int n0   = (wave & 1) * 32;
  const long z   = blockIdx.z;

  const unsigned short* Ab = A + z * sA + (long)(blockIdx.y * 128) * lda;
  const unsigned short* Bb = B + z * sB + (long)(blockIdx.x * 64)  * ldb;
  const long Coff = (z % cMod) * cs1 + (z / cMod) * cs2;

  v8f acc[2][2];
  const v8f vzero = {0.f, 0.f, 0.f, 0.f, 0.f, 0.f, 0.f, 0.f};
  acc[0][0] = vzero; acc[0][1] = vzero; acc[1][0] = vzero; acc[1][1] = vzero;

  const int lm  = lane & 15;
  const int ka  = (lane < 16) ? 0 : 8;    // A-frag K sub-offset (ISA 16-bit A layout)
  const int kb2 = (lane < 16) ? 0 : 8;    // B-frag K sub-offset (ISA 16-bit B layout)
  const int NT  = K >> 5;                 // K tiles of 32

#if HAVE_TDM
  if (wave == 0) {
    tdm_load_tile((unsigned)(uintptr_t)&As[0][0], Ab, lda, 128);
    tdm_load_tile((unsigned)(uintptr_t)&Bs[0][0], Bb, ldb, 64);
  }
#endif

  for (int t = 0; t < NT; ++t) {
    const int buf = t & 1;
#if HAVE_TDM
    if (wave == 0) __builtin_amdgcn_s_wait_tensorcnt(0);  // tile t landed in LDS
    __syncthreads();                                      // visible to all waves
    if (wave == 0 && t + 1 < NT) {                        // overlap DMA(t+1) w/ math(t)
      tdm_load_tile((unsigned)(uintptr_t)&As[buf ^ 1][0], Ab + (t + 1) * 32, lda, 128);
      tdm_load_tile((unsigned)(uintptr_t)&Bs[buf ^ 1][0], Bb + (t + 1) * 32, ldb, 64);
    }
#else
    // fallback: cooperative global->LDS copy
    const int k0 = t * 32;
    #pragma unroll
    for (int i = 0; i < 4; i++) {
      int cidx = tid + i * 256;
      int r = cidx >> 3, cc = (cidx & 7) * 4;
      *(u32x2*)&As[buf][r * 40 + cc] = *(const u32x2*)&Ab[(long)r * lda + k0 + cc];
    }
    #pragma unroll
    for (int i = 0; i < 2; i++) {
      int cidx = tid + i * 256;
      int r = cidx >> 3, cc = (cidx & 7) * 4;
      *(u32x2*)&Bs[buf][r * 40 + cc] = *(const u32x2*)&Bb[(long)r * ldb + k0 + cc];
    }
    __syncthreads();
#endif

    FragAB af[2], bfr[2];
    #pragma unroll
    for (int im = 0; im < 2; im++) {
      int row = m0 + im * 16 + lm;
      af[im].q[0] = *(const u32x4*)&As[buf][row * 40 + ka];        // K = ka..ka+7
      af[im].q[1] = *(const u32x4*)&As[buf][row * 40 + 16 + ka];   // K = 16+ka..
    }
    #pragma unroll
    for (int in = 0; in < 2; in++) {
      int row = n0 + in * 16 + lm;
      bfr[in].q[0] = *(const u32x4*)&Bs[buf][row * 40 + kb2];      // K = kb2..
      bfr[in].q[1] = *(const u32x4*)&Bs[buf][row * 40 + 16 + kb2];
    }
    #pragma unroll
    for (int im = 0; im < 2; im++)
      #pragma unroll
      for (int in = 0; in < 2; in++)
        acc[im][in] = __builtin_amdgcn_wmma_f32_16x16x32_bf16(
            false, af[im].v, false, bfr[in].v, (short)0, acc[im][in], false, false);
#if !HAVE_TDM
    __syncthreads();
#endif
  }

  // epilogue: C/D layout — lanes 0-15: M=vgpr, lanes 16-31: M=8+vgpr; N=lane%16
  const int col   = lane & 15;
  const int rbase = (lane >= 16) ? 8 : 0;
  #pragma unroll
  for (int im = 0; im < 2; im++)
    #pragma unroll
    for (int in = 0; in < 2; in++) {
      int gn = blockIdx.x * 64 + n0 + in * 16 + col;
      float bv = bias ? bias[gn] : 0.0f;
      #pragma unroll
      for (int vr = 0; vr < 8; vr++) {
        int gm = blockIdx.y * 128 + m0 + im * 16 + rbase + vr;
        float val = acc[im][in][vr] + bv;
        long off = Coff + (long)gm * ldc + gn;
        if (out_bf16) ((unsigned short*)C)[off] = f2bf(val);
        else          ((float*)C)[off] = val;
      }
    }
}

// ---------------- bias-included projection -> RoPE + head transpose ----------------
// mode 1: RoPE -> out[b,n,s,d] bf16   (Q, K)
// mode 2: copy -> out[b,n,d,s] bf16   (V transposed for NT attn*V GEMM)
__global__ void rope_transpose_kernel(const float* __restrict__ P,
                                      const int* __restrict__ pos_ids,
                                      unsigned short* __restrict__ out, int mode) {
  int i = blockIdx.x * blockDim.x + threadIdx.x;
  if (i >= BDIM * SEQ * DM) return;
  int c = i & (DM - 1);
  int s = (i >> 10) & (SEQ - 1);
  int b = i >> 20;
  int n = c >> 6;
  int d = c & 63;
  float x = P[i];
  float val = x;
  if (mode == 1) {
    int j = d & 31;
    float pos = (float)pos_ids[b * SEQ + s];
    float ang = pos * __powf(10000.0f, -(float)j * (1.0f / 32.0f));
    float sn, cs;
    __sincosf(ang, &sn, &cs);
    int dp = (d < 32) ? d + 32 : d - 32;
    float xp = P[(i - d) + dp];
    float x2 = (d < 32) ? -xp : xp;
    val = x * cs + x2 * sn;
  }
  long o;
  if (mode == 2) o = (((long)b * NH + n) * DH + d) * SEQ + s;
  else           o = (((long)b * NH + n) * SEQ + s) * DH + d;
  out[o] = f2bf(val);
}

// ---------------- talking-heads pre-mix + mask + softmax + post-mix ----------------
// One block per (q,b) row; all 16 heads processed together (required by head mixing).
// LDS: mix[16][1024] f32 (64KB) + wpre[256] + wpost[256].  In-place bf16 update.
__global__ void __launch_bounds__(256)
talking_softmax_kernel(unsigned short* __restrict__ scores,
                       const float* __restrict__ R,
                       const float* __restrict__ w_pre,
                       const float* __restrict__ w_post) {
  extern __shared__ float smem[];
  float* mix   = smem;                    // 16*1024
  float* wpre  = smem + NH * SEQ;         // 256
  float* wpost = wpre + 256;              // 256

  const int q = blockIdx.x;
  const int b = blockIdx.y;
  const int tid = threadIdx.x;
  wpre[tid]  = w_pre[tid];
  wpost[tid] = w_post[tid];
  __syncthreads();

  // raw + rel score, pre-softmax head mix, scale, causal mask
  #pragma unroll
  for (int i = 0; i < 4; i++) {
    int k = tid + i * 256;
    int idx = q - k;
    idx = idx < -128 ? -128 : (idx > 128 ? 128 : idx);
    idx += 128;
    float raw[NH];
    #pragma unroll
    for (int n = 0; n < NH; n++) {
      long base = ((long)b * NH + n) * SEQ + q;
      raw[n] = bf2f(scores[base * SEQ + k]) + R[base * RELN + idx];
    }
    #pragma unroll
    for (int m = 0; m < NH; m++) {
      float s = 0.f;
      #pragma unroll
      for (int n = 0; n < NH; n++) s += raw[n] * wpre[n * 16 + m];
      s *= 0.125f;                       // / sqrt(d_qk)=8 (linear, commutes with mix)
      if (k > q) s = -10000.0f;
      mix[m * SEQ + k] = s;
    }
  }
  __syncthreads();

  // per-head softmax over k; 8 waves x 2 heads, wave32 shuffle reductions
  const int wave = tid >> 5, lane = tid & 31;
  for (int hh = 0; hh < 2; hh++) {
    int h = wave + hh * 8;
    float mx = -3.4e38f;
    for (int k = lane; k < SEQ; k += 32) mx = fmaxf(mx, mix[h * SEQ + k]);
    #pragma unroll
    for (int off = 16; off > 0; off >>= 1) mx = fmaxf(mx, __shfl_xor(mx, off, 32));
    float sum = 0.f;
    for (int k = lane; k < SEQ; k += 32) {
      float e = __expf(mix[h * SEQ + k] - mx);
      mix[h * SEQ + k] = e;
      sum += e;
    }
    #pragma unroll
    for (int off = 16; off > 0; off >>= 1) sum += __shfl_xor(sum, off, 32);
    float inv = 1.0f / sum;
    for (int k = lane; k < SEQ; k += 32) mix[h * SEQ + k] *= inv;
  }
  __syncthreads();

  // post-softmax head mix, write attn back in place (bf16)
  #pragma unroll
  for (int i = 0; i < 4; i++) {
    int k = tid + i * 256;
    float p[NH];
    #pragma unroll
    for (int m = 0; m < NH; m++) p[m] = mix[m * SEQ + k];
    #pragma unroll
    for (int n = 0; n < NH; n++) {
      float s = 0.f;
      #pragma unroll
      for (int m = 0; m < NH; m++) s += p[m] * wpost[m * 16 + n];
      scores[(((long)b * NH + n) * SEQ + q) * SEQ + k] = f2bf(s);
    }
  }
}

// ---------------- rel-V accumulation: ctx += sum_k attn * rel_v[idx(q,k)] ----------
// Causal ⇒ only idx in [128,256] contribute (masked probs underflow to 0).
// One wave per (b,n,q) row; lanes cover d and d+32.
__global__ void relv_accum_kernel(const unsigned short* __restrict__ attn,
                                  const float* __restrict__ relv,
                                  float* __restrict__ ctx) {
  int gwave = (blockIdx.x * blockDim.x + threadIdx.x) >> 5;
  int lane  = threadIdx.x & 31;
  int q = gwave & (SEQ - 1);
  int n = (gwave >> 10) & (NH - 1);
  int b = gwave >> 14;
  const unsigned short* arow = attn + (((long)b * NH + n) * SEQ + q) * SEQ;
  float acc0 = 0.f, acc1 = 0.f;
  int d0 = lane, d1 = lane + 32;
  int tmax = q < 127 ? q : 127;
  for (int t = 0; t <= tmax; t++) {
    float a = bf2f(arow[q - t]);                   // uniform: broadcast load
    const float* rv = relv + (128 + t) * DH;
    acc0 += a * rv[d0];
    acc1 += a * rv[d1];
  }
  if (q >= 128) {                                  // clipped bin idx==256
    float ts = 0.f;
    for (int k = lane; k <= q - 128; k += 32) ts += bf2f(arow[k]);
    #pragma unroll
    for (int off = 16; off > 0; off >>= 1) ts += __shfl_xor(ts, off, 32);
    acc0 += ts * relv[256 * DH + d0];
    acc1 += ts * relv[256 * DH + d1];
  }
  float* crow = ctx + ((long)b * SEQ + q) * DM + n * DH;
  crow[d0] += acc0;
  crow[d1] += acc1;
}

// ---------------- host-side orchestration ----------------
extern "C" void kernel_launch(void* const* d_in, const int* in_sizes, int n_in,
                              void* d_out, int out_size, void* d_ws, size_t ws_size,
                              hipStream_t stream) {
  (void)in_sizes; (void)n_in; (void)out_size; (void)ws_size;

  const float* query  = (const float*)d_in[0];
  const float* key    = (const float*)d_in[1];
  const float* value  = (const float*)d_in[2];
  // d_in[3] attn_mask: causal triu — handled analytically.
  const int*   q_pos  = (const int*)d_in[4];
  const int*   kv_pos = (const int*)d_in[5];
  const float* W_q = (const float*)d_in[6];   const float* b_q = (const float*)d_in[7];
  const float* W_k = (const float*)d_in[8];   const float* b_k = (const float*)d_in[9];
  const float* W_v = (const float*)d_in[10];  const float* b_v = (const float*)d_in[11];
  const float* W_o = (const float*)d_in[12];  const float* b_o = (const float*)d_in[13];
  const float* relk = (const float*)d_in[14];
  const float* relv = (const float*)d_in[15];
  const float* w_pre  = (const float*)d_in[16];
  const float* w_post = (const float*)d_in[17];
  float* out = (float*)d_out;

  // workspace carve-up (256B aligned)
  char* wsp = (char*)d_ws;
  auto alloc = [&](size_t bytes) {
    char* p = wsp;
    wsp += (bytes + 255) & ~(size_t)255;
    return p;
  };
  const size_t TOK = (size_t)BDIM * SEQ;            // 2048
  unsigned short* qb   = (unsigned short*)alloc(TOK * DM * 2);
  unsigned short* kb   = (unsigned short*)alloc(TOK * DM * 2);
  unsigned short* vb   = (unsigned short*)alloc(TOK * DM * 2);
  unsigned short* wqb  = (unsigned short*)alloc((size_t)DM * DM * 2);
  unsigned short* wkb  = (unsigned short*)alloc((size_t)DM * DM * 2);
  unsigned short* wvb  = (unsigned short*)alloc((size_t)DM * DM * 2);
  unsigned short* wob  = (unsigned short*)alloc((size_t)DM * DM * 2);
  unsigned short* relkp= (unsigned short*)alloc((size_t)RELN * DH * 2);
  float*          P    = (float*)alloc(TOK * DM * 4);
  unsigned short* Qa   = (unsigned short*)alloc((size_t)BDIM * NH * SEQ * DH * 2);
  unsigned short* Ka   = (unsigned short*)alloc((size_t)BDIM * NH * SEQ * DH * 2);
  unsigned short* Vt   = (unsigned short*)alloc((size_t)BDIM * NH * SEQ * DH * 2);
  unsigned short* sc   = (unsigned short*)alloc((size_t)BDIM * NH * SEQ * SEQ * 2); // 64MB, L2-resident
  float*          R    = (float*)alloc((size_t)BDIM * NH * SEQ * RELN * 4);         // 40MB
  float*          ctx  = (float*)alloc(TOK * DM * 4);
  unsigned short* ctxb = (unsigned short*)alloc(TOK * DM * 2);

  const int T = 256;
  const int nTok = (int)(TOK * DM);                 // 2,097,152
  const int nW   = DM * DM;                         // 1,048,576

  // 1) bf16 conversions
  cvt_bf16_kernel<<<(nTok + T - 1) / T, T, 0, stream>>>(query, qb, nTok);
  cvt_bf16_kernel<<<(nTok + T - 1) / T, T, 0, stream>>>(key,   kb, nTok);
  cvt_bf16_kernel<<<(nTok + T - 1) / T, T, 0, stream>>>(value, vb, nTok);
  cvt_bf16_kernel<<<(nW + T - 1) / T, T, 0, stream>>>(W_q, wqb, nW);
  cvt_bf16_kernel<<<(nW + T - 1) / T, T, 0, stream>>>(W_k, wkb, nW);
  cvt_bf16_kernel<<<(nW + T - 1) / T, T, 0, stream>>>(W_v, wvb, nW);
  cvt_bf16_kernel<<<(nW + T - 1) / T, T, 0, stream>>>(W_o, wob, nW);
  pack_relk_kernel<<<(RELN * DH + T - 1) / T, T, 0, stream>>>(relk, relkp);

  // 2) projections (M=2048,N=1024,K=1024) + RoPE / transpose
  gemm_bf16_nt<<<dim3(DM / 64, (int)(TOK / 128), 1), T, 0, stream>>>(
      qb, 0, DM, wqb, 0, DM, P, DM, 1, 0, 0, b_q, DM, 0);
  rope_transpose_kernel<<<(nTok + T - 1) / T, T, 0, stream>>>(P, q_pos, Qa, 1);

  gemm_bf16_nt<<<dim3(DM / 64, (int)(TOK / 128), 1), T, 0, stream>>>(
      kb, 0, DM, wkb, 0, DM, P, DM, 1, 0, 0, b_k, DM, 0);
  rope_transpose_kernel<<<(nTok + T - 1) / T, T, 0, stream>>>(P, kv_pos, Ka, 1);

  gemm_bf16_nt<<<dim3(DM / 64, (int)(TOK / 128), 1), T, 0, stream>>>(
      vb, 0, DM, wvb, 0, DM, P, DM, 1, 0, 0, b_v, DM, 0);
  rope_transpose_kernel<<<(nTok + T - 1) / T, T, 0, stream>>>(P, q_pos, Vt, 2);

  // 3) scores = Q K^T, batched over 32 (b,n), bf16 output (L2-resident)
  gemm_bf16_nt<<<dim3(SEQ / 64, SEQ / 128, BDIM * NH), T, 0, stream>>>(
      Qa, (long)SEQ * DH, DH, Ka, (long)SEQ * DH, DH,
      sc, SEQ, 1, 0, (long)SEQ * SEQ, nullptr, DH, 1);

  // 4) R[bnq, j] = Q . rel_k[j]  (M=32768, N=320, K=64)
  gemm_bf16_nt<<<dim3(RELN / 64, (BDIM * NH * SEQ) / 128, 1), T, 0, stream>>>(
      Qa, 0, DH, relkp, 0, DH, R, RELN, 1, 0, 0, nullptr, DH, 0);

  // 5) talking-heads pre-mix + causal mask + softmax + post-mix (in-place on sc)
  talking_softmax_kernel<<<dim3(SEQ, BDIM), T, (NH * SEQ + 512) * sizeof(float), stream>>>(
      sc, R, w_pre, w_post);

  // 6) ctx = attn @ V (batched, head-interleaved output layout [b,s,n*64+d])
  gemm_bf16_nt<<<dim3(1, SEQ / 128, BDIM * NH), T, 0, stream>>>(
      sc, (long)SEQ * SEQ, SEQ, Vt, (long)DH * SEQ, SEQ,
      ctx, DM, NH, DH, (long)SEQ * DM, nullptr, SEQ, 0);

  // 7) ctx += attn @ rel_v[idx]
  relv_accum_kernel<<<(BDIM * NH * SEQ * 32) / T, T, 0, stream>>>(sc, relv, ctx);

  // 8) output projection: out = ctx @ W_o^T + b_o
  cvt_bf16_kernel<<<(nTok + T - 1) / T, T, 0, stream>>>(ctx, ctxb, nTok);
  gemm_bf16_nt<<<dim3(DM / 64, (int)(TOK / 128), 1), T, 0, stream>>>(
      ctxb, 0, DM, wob, 0, DM, out, DM, 1, 0, 0, b_o, DM, 0);
}
